// HMEClassification_57964878627179
// MI455X (gfx1250) — matrix operand order repaired and checked by reading, not compiled, not run on hardware
//
#include <hip/hip_runtime.h>
#include <hip/hip_bf16.h>

typedef __attribute__((ext_vector_type(16))) __bf16 v16bf;
typedef __attribute__((ext_vector_type(8)))  __bf16 v8bf;
typedef __attribute__((ext_vector_type(8)))  float  v8f;

#define D_DIM 128
#define H_DIM 512
#define C_DIM 64
#define M_TILE 128          // rows per workgroup
#define HC 32               // H chunk width
#define NHC (H_DIM / HC)    // 16 chunks per matrix
#define NSTAGE (7 * NHC)    // 112 total stages (7 matrices x 16 chunks)

// padded LDS strides (bf16 elements) for bank-conflict avoidance
#define XS_S   136
#define W1_S   136
#define W2_S   40
#define HB_S   40

#define W1T_ELEMS (7 * H_DIM * D_DIM)   // 458752 bf16
#define W2T_ELEMS (7 * C_DIM * H_DIM)   // 229376 bf16

__device__ __forceinline__ v16bf frag_cat(const __bf16* p0, const __bf16* p1) {
    v8bf lo = *(const v8bf*)p0;
    v8bf hh = *(const v8bf*)p1;
    v16bf a;
#pragma unroll
    for (int i = 0; i < 8; ++i) { a[i] = lo[i]; a[i + 8] = hh[i]; }
    return a;
}

// CDNA5 async DMA: global -> LDS, 16B per lane, tracked by ASYNCcnt
__device__ __forceinline__ void async_b128(unsigned lds_addr, const void* gptr) {
    asm volatile("global_load_async_to_lds_b128 %0, %1, off"
                 :: "v"(lds_addr), "v"((unsigned long long)(uintptr_t)gptr)
                 : "memory");
}

// ---------------- prep: fp32 weights -> bf16, fragment-friendly layouts ----
// w1T[j][h][d]  (j = 3 gates then 4 experts)
// w2T[j][c][h]  (gate columns c>=2 zero-padded)
__global__ __launch_bounds__(256) void
prep_kernel(const float* __restrict__ gW1, const float* __restrict__ gW2,
            const float* __restrict__ eW1, const float* __restrict__ eW2,
            __bf16* __restrict__ w1T, __bf16* __restrict__ w2T)
{
    int idx0 = blockIdx.x * 256 + threadIdx.x;
    int stride = gridDim.x * 256;
    for (int r = idx0; r < W1T_ELEMS; r += stride) {        // coalesced reads
        int j = r >> 16, rem = r & 65535, d = rem >> 9, h = rem & 511;
        const float* src = (j < 3) ? gW1 + (size_t)j * 65536
                                   : eW1 + (size_t)(j - 3) * 65536;
        w1T[(size_t)j * 65536 + (size_t)h * 128 + d] = (__bf16)src[(size_t)d * 512 + h];
    }
    for (int r = idx0; r < W2T_ELEMS; r += stride) {
        int j = r >> 15, rem = r & 32767, h = rem >> 6, c = rem & 63;
        float v;
        if (j < 3) v = (c < 2) ? gW2[(size_t)j * 1024 + h * 2 + c] : 0.f;
        else       v = eW2[(size_t)(j - 3) * 32768 + (size_t)h * 64 + c];
        w2T[(size_t)j * 32768 + (size_t)c * 512 + h] = (__bf16)v;
    }
}

// ---------------- main fused HME kernel ----------------
struct StageBufs {
    __bf16 w1[3][HC][W1_S];      // 3 x 8704 B
    __bf16 w2[3][C_DIM][W2_S];   // 3 x 5120 B
};
union Smem {
    __bf16 xs[M_TILE][XS_S];     // 34816 B (only live before main loop)
    StageBufs st;                // 41472 B (triple-buffered weight slabs)
};

__global__ __launch_bounds__(256) void
hme_kernel(const float* __restrict__ x,
           const float* __restrict__ gb1, const float* __restrict__ gb2,
           const float* __restrict__ eb1, const float* __restrict__ eb2,
           const __bf16* __restrict__ w1T, const __bf16* __restrict__ w2T,
           float* __restrict__ out)
{
    __shared__ Smem  sm;
    __shared__ __bf16 hbuf[8][16][HB_S];   // 10240 B per-wave hidden chunk
    __shared__ float  gatebuf[M_TILE][6];  //  3072 B
    __shared__ float  coefbuf[M_TILE][4];  //  2048 B

    const int tid  = threadIdx.x;
    const int wv   = tid >> 5;
    const int lane = tid & 31;
    const int l16  = lane & 15;
    const int hi   = lane >> 4;
    const int row0 = blockIdx.x * M_TILE;

    // ---- stage x tile as bf16 (once) ----
    for (int idx = tid; idx < M_TILE * D_DIM; idx += 256) {
        int r = idx >> 7, d = idx & 127;
        sm.xs[r][d] = (__bf16)x[(size_t)(row0 + r) * D_DIM + d];
    }
    __syncthreads();

    // ---- persistent x A-fragments (16 rows x 32 K each) ----
    v16bf afr[4];
#pragma unroll
    for (int kk = 0; kk < 4; ++kk) {
        int row = wv * 16 + l16;
        int klo = kk * 32 + (hi ? 8 : 0);
        afr[kk] = frag_cat(&sm.xs[row][klo], &sm.xs[row][klo + 16]);
    }
    __syncthreads();   // xs dead; union memory may now be overwritten by DMA

    const unsigned w1b0 = (unsigned)(uintptr_t)&sm.st.w1[0][0][0];
    const unsigned w2b0 = (unsigned)(uintptr_t)&sm.st.w2[0][0][0];

    // issue one stage (3 async ops per thread): W1 slab 8KB + W2 slab 4KB
    auto issue = [&](int t, int b) {
        int jn = t >> 4, hcn = t & 15;
        const __bf16* g1 = w1T + ((size_t)jn * H_DIM + (size_t)hcn * HC) * D_DIM;
        unsigned l1 = w1b0 + (unsigned)b * (HC * W1_S * 2);
#pragma unroll
        for (int i = 0; i < 2; ++i) {
            int q = tid + i * 256;          // 512 x 16B chunks
            int hl = q >> 4, part = q & 15;
            async_b128(l1 + (unsigned)(hl * W1_S + part * 8) * 2,
                       g1 + hl * D_DIM + part * 8);
        }
        int c = tid >> 2, part = tid & 3;   // 256 x 16B chunks
        const __bf16* g2 = w2T + ((size_t)jn * C_DIM + c) * H_DIM
                               + (size_t)hcn * HC + part * 8;
        async_b128(w2b0 + (unsigned)b * (C_DIM * W2_S * 2)
                        + (unsigned)(c * W2_S + part * 8) * 2, g2);
    };

    issue(0, 0);   // prologue

    v8f outacc[4];
#pragma unroll
    for (int t = 0; t < 4; ++t)
#pragma unroll
        for (int i = 0; i < 8; ++i) outacc[t][i] = 0.f;

#pragma unroll 1
    for (int j = 0; j < 7; ++j) {
        const bool gate = (j < 3);
        const int  e    = j - 3;
        const float* B1 = gate ? gb1 + j * H_DIM : eb1 + e * H_DIM;
        const float* B2 = gate ? gb2 + j * 2     : eb2 + e * C_DIM;
        const int NT2 = gate ? 1 : 4;

        v8f acc2[4];
#pragma unroll
        for (int t = 0; t < 4; ++t)
#pragma unroll
            for (int i = 0; i < 8; ++i) acc2[t][i] = 0.f;

#pragma unroll 1
        for (int hc = 0; hc < NHC; ++hc) {
            const int t = j * NHC + hc;
            const int b = t % 3;
            if (t + 1 < NSTAGE) {
                issue(t + 1, (t + 1) % 3);               // overlap DMA w/ compute
                asm volatile("s_wait_asynccnt 3" ::: "memory"); // stage t landed
            } else {
                asm volatile("s_wait_asynccnt 0" ::: "memory");
            }
            __syncthreads();

            const __bf16 (*w1s)[W1_S] = sm.st.w1[b];
            const __bf16 (*w2s)[W2_S] = sm.st.w2[b];

            // ---- GEMM1: h = relu(x @ W1 + b1), 16x32 per wave ----
            v8f acc1[2];
#pragma unroll
            for (int tt = 0; tt < 2; ++tt)
#pragma unroll
                for (int i = 0; i < 8; ++i) acc1[tt][i] = 0.f;
#pragma unroll
            for (int half = 0; half < 2; ++half) {
                v16bf bf[4];
#pragma unroll
                for (int kk2 = 0; kk2 < 2; ++kk2)
#pragma unroll
                    for (int nt = 0; nt < 2; ++nt) {
                        const __bf16* bp =
                            &w1s[nt * 16 + l16][(half * 2 + kk2) * 32 + (hi ? 8 : 0)];
                        bf[kk2 * 2 + nt] = frag_cat(bp, bp + 16);
                    }
#pragma unroll
                for (int kk2 = 0; kk2 < 2; ++kk2)
#pragma unroll
                    for (int nt = 0; nt < 2; ++nt)
                        acc1[nt] = __builtin_amdgcn_wmma_f32_16x16x32_bf16(
                            false, afr[half * 2 + kk2], false, bf[kk2 * 2 + nt],
                            (short)0, acc1[nt], false, false);
            }

            // preload GEMM2 B fragments (independent of hbuf round-trip)
            v16bf b2f[4];
#pragma unroll
            for (int nt2 = 0; nt2 < 4; ++nt2) {
                if (nt2 >= NT2) break;
                const __bf16* bp = &w2s[nt2 * 16 + l16][hi ? 8 : 0];
                b2f[nt2] = frag_cat(bp, bp + 16);
            }

            // bias + relu, C-layout -> per-wave LDS as bf16
#pragma unroll
            for (int nt = 0; nt < 2; ++nt) {
                float bi = B1[hc * HC + nt * 16 + l16];
#pragma unroll
                for (int r = 0; r < 8; ++r) {
                    float v = acc1[nt][r] + bi;
                    v = v > 0.f ? v : 0.f;
                    hbuf[wv][r + 8 * hi][nt * 16 + l16] = (__bf16)v;
                }
            }
            asm volatile("s_wait_dscnt 0" ::: "memory");   // same-wave LDS RAW

            int klo = hi ? 8 : 0;
            v16bf a2 = frag_cat(&hbuf[wv][l16][klo], &hbuf[wv][l16][klo + 16]);
#pragma unroll
            for (int nt2 = 0; nt2 < 4; ++nt2) {
                if (nt2 >= NT2) break;
                acc2[nt2] = __builtin_amdgcn_wmma_f32_16x16x32_bf16(
                    false, a2, false, b2f[nt2], (short)0, acc2[nt2], false, false);
            }
        } // hc

        if (gate) {
            float bi = (l16 < 2) ? B2[l16] : 0.f;
#pragma unroll
            for (int r = 0; r < 8; ++r) {
                float v = acc2[0][r] + bi;
                float o = __shfl_xor(v, 1, 32);
                float m = fmaxf(v, o);
                float ev = __expf(v - m), eo = __expf(o - m);
                float p = ev / (ev + eo);
                if (l16 < 2) gatebuf[wv * 16 + r + 8 * hi][j * 2 + l16] = p;
            }
            if (j == 2) {
                asm volatile("s_wait_dscnt 0" ::: "memory");
                if (lane < 16) {
                    int row = wv * 16 + lane;
                    float r0 = gatebuf[row][0], r1 = gatebuf[row][1];
                    float a0 = gatebuf[row][2], a1 = gatebuf[row][3];
                    float b0 = gatebuf[row][4], b1 = gatebuf[row][5];
                    coefbuf[row][0] = r0 * a0;
                    coefbuf[row][1] = r0 * a1;
                    coefbuf[row][2] = r1 * b0;
                    coefbuf[row][3] = r1 * b1;
                }
                asm volatile("s_wait_dscnt 0" ::: "memory");
            }
        } else {
            float bi[4];
#pragma unroll
            for (int t = 0; t < 4; ++t) bi[t] = B2[t * 16 + l16];
#pragma unroll
            for (int r = 0; r < 8; ++r) {
                float l0 = acc2[0][r] + bi[0], l1 = acc2[1][r] + bi[1];
                float l2 = acc2[2][r] + bi[2], l3 = acc2[3][r] + bi[3];
                float m = fmaxf(fmaxf(l0, l1), fmaxf(l2, l3));
#pragma unroll
                for (int s = 1; s < 16; s <<= 1) m = fmaxf(m, __shfl_xor(m, s, 32));
                float e0 = __expf(l0 - m), e1 = __expf(l1 - m);
                float e2 = __expf(l2 - m), e3 = __expf(l3 - m);
                float sm2 = e0 + e1 + e2 + e3;
#pragma unroll
                for (int s = 1; s < 16; s <<= 1) sm2 += __shfl_xor(sm2, s, 32);
                float cf = coefbuf[wv * 16 + r + 8 * hi][e] / sm2;
                outacc[0][r] += cf * e0;
                outacc[1][r] += cf * e1;
                outacc[2][r] += cf * e2;
                outacc[3][r] += cf * e3;
            }
        }
    } // j

    // ---- coalesced output store ----
#pragma unroll
    for (int nt2 = 0; nt2 < 4; ++nt2)
#pragma unroll
        for (int r = 0; r < 8; ++r)
            out[(size_t)(row0 + wv * 16 + r + 8 * hi) * C_DIM + nt2 * 16 + l16] =
                outacc[nt2][r];
}

extern "C" void kernel_launch(void* const* d_in, const int* in_sizes, int n_in,
                              void* d_out, int out_size, void* d_ws, size_t ws_size,
                              hipStream_t stream) {
    const float* x   = (const float*)d_in[0];
    const float* gW1 = (const float*)d_in[1];
    const float* gb1 = (const float*)d_in[2];
    const float* gW2 = (const float*)d_in[3];
    const float* gb2 = (const float*)d_in[4];
    const float* eW1 = (const float*)d_in[5];
    const float* eb1 = (const float*)d_in[6];
    const float* eW2 = (const float*)d_in[7];
    const float* eb2 = (const float*)d_in[8];
    float* out = (float*)d_out;

    __bf16* w1T = (__bf16*)d_ws;
    __bf16* w2T = w1T + W1T_ELEMS;     // needs ~1.32 MB of d_ws

    prep_kernel<<<dim3(512), dim3(256), 0, stream>>>(gW1, gW2, eW1, eW2, w1T, w2T);

    int B = in_sizes[0] / D_DIM;       // 131072
    hme_kernel<<<dim3(B / M_TILE), dim3(256), 0, stream>>>(
        x, gb1, gb2, eb1, eb2, w1T, w2T, out);
}
